// DeformableAttnBlock_FUSION_57887569216155
// MI455X (gfx1250) — matrix operand order repaired and checked by reading, not compiled, not run on hardware
//
#include <hip/hip_runtime.h>
#include <hip/hip_bf16.h>
#include <cstdint>

#define BB 2
#define TT 3
#define CC 64
#define HH 256
#define WW 256
#define NN (HH*WW)
#define NHEADS 4
#define HEAD_DIM 16
#define NLEV 3
#define NPTS 4
#define NPATH 5

typedef __bf16 bf16;
typedef __attribute__((ext_vector_type(16))) __bf16 v16bf;
typedef __attribute__((ext_vector_type(8)))  float  v8f;
typedef __attribute__((ext_vector_type(4)))  unsigned int v4u;
typedef __attribute__((ext_vector_type(8)))  int v8i;
typedef __attribute__((ext_vector_type(4)))  int v4i;

__device__ __forceinline__ float lrelu(float x){ return x >= 0.f ? x : 0.1f*x; }

__device__ __forceinline__ v8f wmma_bf16(v16bf a, v16bf b, v8f c){
  return __builtin_amdgcn_wmma_f32_16x16x32_bf16(false, a, false, b, (short)0, c, false, false);
}

// A-fragment: two contiguous 8-element (16B) runs per lane (callers pass adjacent runs: k = 16*(lane/16) + s)
__device__ __forceinline__ v16bf load_frag_pair(const bf16* p0, const bf16* p1){
  v16bf r;
  uint4* u = reinterpret_cast<uint4*>(&r);
  u[0] = *reinterpret_cast<const uint4*>(p0);
  u[1] = *reinterpret_cast<const uint4*>(p1);
  return r;
}
// B-fragment: one contiguous 16-element (32B) run per lane (Wt pre-transposed [n][k])
__device__ __forceinline__ v16bf load_frag32(const bf16* p){
  v16bf r;
  uint4* u = reinterpret_cast<uint4*>(&r);
  const uint4* s = reinterpret_cast<const uint4*>(p);
  u[0] = s[0]; u[1] = s[1];
  return r;
}

// CDNA5 async global->LDS copy (ASYNCcnt-tracked, no VGPR data path)
__device__ __forceinline__ void async_load_b128(uint32_t lds, const void* gaddr){
  asm volatile("global_load_async_to_lds_b128 %0, %1, off"
               :: "v"(lds), "v"(gaddr) : "memory");
}
__device__ __forceinline__ void wait_async0(){ asm volatile("s_wait_asynccnt 0x0" ::: "memory"); }
__device__ __forceinline__ void wait_async2(){ asm volatile("s_wait_asynccnt 0x2" ::: "memory"); }

// CDNA5 Tensor Data Mover: 2D tile (tile_d1 rows x tile_d0 elems, 2-byte data,
// row stride = stride_elems) from global -> contiguous LDS. D# per ISA ch.8.
__device__ __forceinline__ void tdm_load_2d(uint32_t lds_addr, const void* gaddr,
                                            uint32_t dims_d0, uint32_t dims_d1,
                                            uint32_t tile_d0, uint32_t tile_d1,
                                            uint32_t stride_elems){
  uint64_t ga = (uint64_t)(uintptr_t)gaddr;
  v4u g0 = { 1u,                                   // count=1, user descriptor
             lds_addr,                             // LDS byte address
             (uint32_t)ga,                         // global_addr[31:0]
             (uint32_t)((ga>>32) & 0x01FFFFFFu) | 0x80000000u };  // addr[56:32] | type=2
  v8i g1 = { (int)(1u<<16),                        // data_size=1 -> 2 bytes
             (int)((dims_d0 & 0xffffu) << 16),     // tensor_dim0[15:0] in [31:16]
             (int)((dims_d0 >> 16) | ((dims_d1 & 0xffffu) << 16)),
             (int)((dims_d1 >> 16) | (tile_d0 << 16)),
             (int)(tile_d1 & 0xffffu),             // tile_dim1 ([15:0]), tile_dim2=0
             (int)stride_elems,                    // tensor_dim0_stride[31:0]
             0, 0 };                               // stride hi, dim1_stride = 0
  v4i z4 = {0,0,0,0};
#if __clang_major__ >= 23
  v8i z8 = {0,0,0,0,0,0,0,0};
  __builtin_amdgcn_tensor_load_to_lds(g0, g1, z4, z4, z8, 0);
#else
  __builtin_amdgcn_tensor_load_to_lds(g0, g1, z4, z4, 0);
#endif
}

// ---------------------------------------------------------------------------
// Weight prepack: fp32 -> bf16, layouts chosen so WMMA B-fragments are
// contiguous 32B lane loads. Convs: [oc][kk][icp] (icp zero-padded to mult 32).
// GEMMs: Wt[n][k] (K contiguous).
// ---------------------------------------------------------------------------
__global__ __launch_bounds__(256)
void k_prepack(const float* __restrict__ eqkw, const float* __restrict__ evw,
               const float* __restrict__ ffnw, const float* __restrict__ vprojw,
               const float* __restrict__ offw, const float* __restrict__ aww,
               const float* __restrict__ fusw,
               bf16* __restrict__ wqk, bf16* __restrict__ wv, bf16* __restrict__ wffn,
               bf16* __restrict__ wtv, bf16* __restrict__ wto, bf16* __restrict__ wta,
               bf16* __restrict__ wtf)
{
  int idx = blockIdx.x*256 + threadIdx.x;
  const int n1=64*9*224, n2=192*9*192, n3=64*9*64, n4=64*64, n5=96*64, n6=48*64, n7=64*64;
  if(idx < n1){
    int ocl=idx/(9*224); int r=idx%(9*224); int kk=r/224; int ic=r%224;
    wqk[idx] = (bf16)((ic<196)? eqkw[((size_t)(64+ocl)*196+ic)*9+kk] : 0.f);
  } else if((idx-=n1) < n2){
    int oc=idx/(9*192); int kk=(idx/192)%9; int ic=idx%192;
    wv[idx] = (bf16)evw[((size_t)oc*192+ic)*9+kk];
  } else if((idx-=n2) < n3){
    int oc=idx/(9*64); int kk=(idx/64)%9; int ic=idx%64;
    wffn[idx] = (bf16)ffnw[((size_t)oc*64+ic)*9+kk];
  } else if((idx-=n3) < n4){
    int n=idx/64, k=idx%64; wtv[idx] = (bf16)vprojw[k*64+n];
  } else if((idx-=n4) < n5){
    int n=idx/64, k=idx%64; wto[idx] = (bf16)offw[k*96+n];
  } else if((idx-=n5) < n6){
    int n=idx/64, k=idx%64; wta[idx] = (bf16)aww[k*48+n];
  } else if((idx-=n6) < n7){
    wtf[idx] = (bf16)fusw[idx];   // [oc][ic] already == Wt[n][k]
  }
}

// ---------------------------------------------------------------------------
// gate = mean(frame_in, HW) + prior   (one block per (b,c))
// ---------------------------------------------------------------------------
__global__ __launch_bounds__(256)
void k_gate_mean(const float* __restrict__ frame_in, const float* __restrict__ prior,
                 float* __restrict__ gate)
{
  int bc = blockIdx.x;
  const float* src = frame_in + (size_t)bc*NN;
  float s = 0.f;
  for(int i=threadIdx.x; i<NN; i+=256) s += src[i];
  __shared__ float red[256];
  red[threadIdx.x]=s; __syncthreads();
  for(int st=128; st>0; st>>=1){
    if(threadIdx.x<st) red[threadIdx.x]+=red[threadIdx.x+st];
    __syncthreads();
  }
  if(threadIdx.x==0) gate[bc] = red[0]/(float)NN + prior[bc];
}

// threefry2x32 (JAX-style, key=(0,42))
__device__ __forceinline__ uint32_t rotl32(uint32_t x,int r){ return (x<<r)|(x>>(32-r)); }
__device__ void tf2x32(uint32_t k0,uint32_t k1,uint32_t x0,uint32_t x1,uint32_t&o0,uint32_t&o1){
  uint32_t ks[3]={k0,k1,k0^k1^0x1BD11BDAu};
  const int R0[4]={13,15,26,6}; const int R1[4]={17,29,16,24};
  x0+=ks[0]; x1+=ks[1];
  for(int i=0;i<5;i++){
    const int* R = (i&1)?R1:R0;
    for(int r=0;r<4;r++){ x0+=x1; x1=rotl32(x1,R[r]); x1^=x0; }
    x0+=ks[(i+1)%3]; x1+=ks[(i+2)%3]+(uint32_t)(i+1);
  }
  o0=x0; o1=x1;
}

// ---------------------------------------------------------------------------
// select (gumbel-softmax straight-through -> one-hot), then fold 5-path
// outproj into per-batch W_eff / b_eff. Wt_eff stored transposed [b][n][k] bf16.
// ---------------------------------------------------------------------------
__global__ __launch_bounds__(256)
void k_select(const float* __restrict__ gate, const float* __restrict__ map_w,
              const float* __restrict__ map_b, const float* __restrict__ outproj_w,
              const float* __restrict__ outproj_b,
              bf16* __restrict__ wteff, float* __restrict__ beff)
{
  __shared__ float ssel[BB*NPATH];
  if(threadIdx.x==0){
    uint32_t ubits[10];
    for(int i=0;i<5;i++){
      uint32_t a,b2; tf2x32(0u,42u,(uint32_t)i,(uint32_t)(5+i),a,b2);
      ubits[i]=a; ubits[5+i]=b2;
    }
    for(int b=0;b<BB;b++){
      float logit[NPATH], z[NPATH];
      for(int p=0;p<NPATH;p++){
        float s=map_b[p];
        for(int c=0;c<CC;c++) s += map_w[p*CC+c]*gate[b*CC+c];
        logit[p]=s;
      }
      float zmax=-1e30f;
      for(int p=0;p<NPATH;p++){
        uint32_t bits = ubits[b*NPATH+p];
        float u = __uint_as_float((bits>>9)|0x3f800000u) - 1.0f;   // [0,1)
        u = 1e-10f + u*(1.0f - 1e-10f);
        float gum = -logf(-logf(u));
        z[p] = (logit[p]+gum)*(1.0f/0.1f);
        zmax = fmaxf(zmax, z[p]);
      }
      float se=0.f; int amax=0; float smax=-1e30f;
      float soft[NPATH];
      for(int p=0;p<NPATH;p++){ soft[p]=__expf(z[p]-zmax); se+=soft[p]; }
      for(int p=0;p<NPATH;p++){
        soft[p]/=se;
        if(soft[p]>smax){ smax=soft[p]; amax=p; }
      }
      for(int p=0;p<NPATH;p++){
        float hard = (p==amax)?1.f:0.f;
        ssel[b*NPATH+p] = (hard + soft[p]) - soft[p];   // straight-through forward
      }
    }
  }
  __syncthreads();
  for(int idx=threadIdx.x; idx<BB*CC*CC; idx+=256){
    int b=idx/(CC*CC); int r=idx%(CC*CC); int n=r/CC; int k=r%CC;
    float s=0.f;
    for(int p=0;p<NPATH;p++) s += ssel[b*NPATH+p]*outproj_w[((size_t)p*CC+k)*CC+n];
    wteff[idx] = (bf16)s;
  }
  for(int idx=threadIdx.x; idx<BB*CC; idx+=256){
    int b=idx/CC, n=idx%CC; float s=0.f;
    for(int p=0;p<NPATH;p++) s += ssel[b*NPATH+p]*outproj_b[p*CC+n];
    beff[idx]=s;
  }
}

// ---------------------------------------------------------------------------
// Stage: border-clamped bilinear warps + concat -> qk_in (NHWC, 224ch padded,
// bf16) and frame NHWC 192ch bf16 for emb_v. One thread per (b,pixel).
// ---------------------------------------------------------------------------
__global__ __launch_bounds__(256)
void k_stage(const float* __restrict__ frame_in, const float* __restrict__ srcframe,
             const float* __restrict__ flow_f, const float* __restrict__ flow_b,
             bf16* __restrict__ qkin, bf16* __restrict__ fr192)
{
  size_t id = (size_t)blockIdx.x*256 + threadIdx.x;
  if(id >= (size_t)BB*NN) return;
  int b = (int)(id / NN); int hw = (int)(id % NN);
  int y = hw >> 8, x = hw & 255;
  float fbx = flow_b[(((size_t)b*2+0)*2+0)*NN + hw];
  float fby = flow_b[(((size_t)b*2+0)*2+1)*NN + hw];
  float ffx = flow_f[(((size_t)b*2+1)*2+0)*NN + hw];
  float ffy = flow_f[(((size_t)b*2+1)*2+1)*NN + hw];
  bf16* qrow = qkin  + id*224;
  bf16* frow = fr192 + id*192;
  #pragma unroll
  for(int srci=0; srci<2; srci++){
    float gx = (float)x + (srci==0?fbx:ffx);
    float gy = (float)y + (srci==0?fby:ffy);
    gx = fminf(fmaxf(gx,0.f), (float)(WW-1));
    gy = fminf(fmaxf(gy,0.f), (float)(HH-1));
    float x0f=floorf(gx), y0f=floorf(gy);
    float wx=gx-x0f, wy=gy-y0f;
    int x0i=(int)x0f, y0i=(int)y0f;
    int x1i=min(x0i+1, WW-1), y1i=min(y0i+1, HH-1);
    float w00=(1.f-wx)*(1.f-wy), w01=wx*(1.f-wy), w10=(1.f-wx)*wy, w11=wx*wy;
    int t=(srci==0)?0:2;
    const float* base = srcframe + ((size_t)(b*TT+t))*CC*NN;
    size_t p00=(size_t)y0i*WW+x0i, p01=(size_t)y0i*WW+x1i;
    size_t p10=(size_t)y1i*WW+x0i, p11=(size_t)y1i*WW+x1i;
    for(int c=0;c<CC;c++){
      const float* bc = base + (size_t)c*NN;
      float v = w00*bc[p00]+w01*bc[p01]+w10*bc[p10]+w11*bc[p11];
      qrow[srci*128 + c] = (bf16)v;
    }
  }
  for(int c=0;c<CC;c++){
    float f1 = frame_in[((size_t)(b*CC+c))*NN + hw];
    qrow[64+c] = (bf16)f1;
    frow[64+c] = (bf16)f1;
    frow[c]      = (bf16)srcframe[(((size_t)(b*TT+0))*CC + c)*NN + hw];
    frow[128+c]  = (bf16)srcframe[(((size_t)(b*TT+2))*CC + c)*NN + hw];
  }
  qrow[192]=(bf16)ffx; qrow[193]=(bf16)ffy; qrow[194]=(bf16)fbx; qrow[195]=(bf16)fby;
  #pragma unroll
  for(int c=196;c<224;c++) qrow[c]=(bf16)0.f;
}

// ---------------------------------------------------------------------------
// Implicit-GEMM 3x3 conv on WMMA bf16. Block = 128 pixels x 64 oc, 8 waves.
// A tile double-buffered in LDS:
//   - interior tiles: one TDM descriptor (tensor_load_to_lds) issued by wave 0,
//     TENSORcnt-waited (2D tile: 128 rows x 32 ch, row stride ICP)
//   - halo tiles: per-lane global_load_async_to_lds_b128 with zero-page
//     redirect (exactly 2 async instrs/wave/chunk -> deterministic waits)
// MODE 0: emb_qk (q slice, +bias[64+n], lrelu) -> q NHWC bf16
// MODE 1: emb_v  (+bias, lrelu)                -> valA [(b,t,hw)][c] bf16
// MODE 2: ffn    (+bias, +frame1)              -> fusin NHWC bf16
// ---------------------------------------------------------------------------
template<int ICP, int MODE>
__global__ __launch_bounds__(256)
void k_conv3x3(const bf16* __restrict__ in, const bf16* __restrict__ wpk,
               const float* __restrict__ bias, bf16* __restrict__ out,
               const float* __restrict__ aux, const char* __restrict__ zpad)
{
  constexpr int NCH = ICP/32;      // 32-wide K chunks per (ky,kx)
  constexpr int NC  = 9*NCH;       // total chunks
  __shared__ bf16 sA[2][128*32];
  const int tile = blockIdx.x;
  const int ocb  = blockIdx.y;
  const int x0   = (tile & 1) * 128;
  const int y    = (tile >> 1) & (HH-1);
  const int b    = tile >> 9;
  const int lane = threadIdx.x & 31;
  const int wave = threadIdx.x >> 5;
  const int colc = lane & 15;
  const int kseg = lane >> 4;
  const int o0   = (lane < 16) ? 0 : 8;
  const int arow = wave*16 + colc;
  const int srow = threadIdx.x >> 1;
  const int shalf= (threadIdx.x & 1) * 16;
  const uint32_t ldsBuf0   = (uint32_t)(uintptr_t)(&sA[0][0]);
  const uint32_t ldsStageOff = (uint32_t)(srow*32 + shalf)*2u;

  v8f z = {0,0,0,0,0,0,0,0};
  v8f acc[4];
  #pragma unroll
  for(int i=0;i<4;i++) acc[i]=z;

  const bf16* wcol[4];
  #pragma unroll
  for(int nt=0; nt<4; nt++)
    wcol[nt] = wpk + ((size_t)(ocb*64 + nt*16 + colc)*9)*ICP + kseg*16;

  // returns true if the chunk was issued via TDM (interior tile)
  auto issue = [&](int ci, int buf)->bool{
    const int kk = ci / NCH, cc = (ci % NCH)*32;
    const int ky = kk/3 - 1, kx = kk%3 - 1;
    const int yy = y + ky;
    const int xs = x0 + kx;
    const bool interior = (yy>=0 && yy<HH && xs>=0 && xs+128<=WW);
    const uint32_t lbase = ldsBuf0 + (uint32_t)buf*(128*32*2);
    if(interior){
      if(wave == 0){
        const void* g = (const void*)(in + ((size_t)((b*HH+yy)*WW+xs))*ICP + cc);
        tdm_load_2d(lbase, g, /*dims*/32u, 128u, /*tile*/32u, 128u, (uint32_t)ICP);
      }
      return true;
    }
    const int xx = xs + srow;
    const bool valid = (yy>=0 && yy<HH && xx>=0 && xx<WW);
    const char* g = valid
        ? (const char*)(in + ((size_t)((b*HH+yy)*WW+xx))*ICP + cc + shalf)
        : zpad;
    async_load_b128(lbase + ldsStageOff,      g);
    async_load_b128(lbase + ldsStageOff + 16, g + (valid ? 16 : 0));
    return false;
  };

  bool curT = issue(0, 0);
  for(int ci=0; ci<NC; ci++){
    const int buf = ci & 1;
    const bool haveNext = (ci+1 < NC);
    bool nextT = false;
    if(haveNext) nextT = issue(ci+1, buf^1);
    // wait for the CURRENT buffer's copies (in-order per counter class)
    if(curT){
      if(wave == 0){
        if(haveNext && nextT) __builtin_amdgcn_s_wait_tensorcnt((short)1);
        else                  __builtin_amdgcn_s_wait_tensorcnt((short)0);
      }
    } else {
      if(haveNext && !nextT) wait_async2();
      else                   wait_async0();
    }
    __syncthreads();
    const int kk = ci / NCH, cc = (ci % NCH)*32;
    const bf16* sbuf = &sA[buf][0];
    v16bf afr = load_frag_pair(sbuf + arow*32 + kseg*16, sbuf + arow*32 + kseg*16 + 8);
    v16bf bfr[4];
    #pragma unroll
    for(int nt=0; nt<4; nt++) bfr[nt] = load_frag32(wcol[nt] + (size_t)kk*ICP + cc);
    #pragma unroll
    for(int nt=0; nt<4; nt++) acc[nt] = wmma_bf16(afr, bfr[nt], acc[nt]);
    __syncthreads();
    curT = nextT;
  }

  const int hwbase = y*WW + x0;
  #pragma unroll
  for(int nt=0; nt<4; nt++){
    const int n = ocb*64 + nt*16 + colc;
    #pragma unroll
    for(int v=0; v<8; v++){
      const int row = wave*16 + v + kseg*8;
      const int hw  = hwbase + row;
      float val = acc[nt][v];
      if(MODE==0){
        val = lrelu(val + bias[64 + n]);
        out[((size_t)(b*NN + hw))*64 + n] = (bf16)val;
      } else if(MODE==1){
        val = lrelu(val + bias[n]);
        const int t = ocb; const int c = n - ocb*64;
        out[(((size_t)(b*TT + t))*NN + hw)*64 + c] = (bf16)val;
      } else {
        val = val + bias[n] + aux[((size_t)(b*64 + n))*NN + hw];
        out[((size_t)(b*NN + hw))*64 + n] = (bf16)val;
      }
    }
  }
}

// ---------------------------------------------------------------------------
// WMMA GEMM, K=64, N = NB*16. A bf16 row-major, Wt bf16 [n][k].
// All B fragments loaded before the WMMA chain (single wait, no serialization).
// MODE 10: vproj  -> vsamp bf16 [(b,t,head,hw)][d]
// MODE 11: off logits -> f32 [m][96]
// MODE 12: aw  logits -> f32 [m][48]
// MODE 13: outproj (per-batch W_eff/b_eff) -> ffnin NHWC bf16
// MODE 14: fus 1x1 + lrelu -> d_out f32 NCHW
// ---------------------------------------------------------------------------
template<int NB, int MODE>
__global__ __launch_bounds__(256)
void k_gemm64(const bf16* __restrict__ A, const bf16* __restrict__ Wt,
              const float* __restrict__ bias, void* __restrict__ outv)
{
  const int lane = threadIdx.x & 31;
  const int wave = threadIdx.x >> 5;
  const int colc = lane & 15;
  const int kseg = lane >> 4;
  const int o0   = (lane<16)?0:8;
  const size_t rowa = (size_t)blockIdx.x*128 + wave*16 + colc;

  v8f z = {0,0,0,0,0,0,0,0};
  v8f acc[NB];
  #pragma unroll
  for(int i=0;i<NB;i++) acc[i]=z;

  const bf16* arow = A + rowa*64;
  const bf16* wt = Wt;
  if(MODE==13) wt += ((size_t)blockIdx.x*128/NN)*(64*64);
  __builtin_prefetch(arow + 128*64, 0, 0);         // next M-tile
  #pragma unroll
  for(int ck=0; ck<2; ck++){
    v16bf bfr[NB];
    #pragma unroll
    for(int nt=0; nt<NB; nt++)
      bfr[nt] = load_frag32(wt + (size_t)(nt*16+colc)*64 + ck*32 + kseg*16);
    v16bf afr = load_frag_pair(arow + ck*32 + kseg*16, arow + ck*32 + kseg*16 + 8);
    #pragma unroll
    for(int nt=0; nt<NB; nt++) acc[nt] = wmma_bf16(afr, bfr[nt], acc[nt]);
  }
  #pragma unroll
  for(int nt=0; nt<NB; nt++){
    const int n = nt*16 + colc;
    #pragma unroll
    for(int v=0; v<8; v++){
      const size_t m = (size_t)blockIdx.x*128 + wave*16 + v + kseg*8;
      float val = acc[nt][v];
      if(MODE==10){
        val += bias[n];
        size_t b=m/(TT*NN); size_t r=m%(TT*NN); size_t t=r/NN; size_t hw=r%NN;
        ((bf16*)outv)[((((b*TT+t)*NHEADS + (size_t)(n>>4))*NN) + hw)*HEAD_DIM + (n&15)] = (bf16)val;
      } else if(MODE==11){
        ((float*)outv)[m*96+n] = val + bias[n];
      } else if(MODE==12){
        ((float*)outv)[m*48+n] = val + bias[n];
      } else if(MODE==13){
        size_t b=m/NN;
        ((bf16*)outv)[m*64+n] = (bf16)(val + bias[b*64+n]);
      } else {
        size_t b=m/NN, hw=m%NN;
        ((float*)outv)[(b*64+n)*NN+hw] = lrelu(val + bias[n]);
      }
    }
  }
}

// ---------------------------------------------------------------------------
// Deformable sampling: one thread per (b,pixel,head); softmax(12) over aw,
// 12 zero-padded bilinear taps into vsamp (16 contiguous bf16 per pixel).
// ---------------------------------------------------------------------------
__global__ __launch_bounds__(256)
void k_sample(const float* __restrict__ offb, const float* __restrict__ awb,
              const bf16* __restrict__ vsamp, const float* __restrict__ flow_f,
              const float* __restrict__ flow_b, bf16* __restrict__ attn)
{
  size_t id = (size_t)blockIdx.x*256 + threadIdx.x;
  if(id >= (size_t)BB*NN*NHEADS) return;
  int head = (int)(id & 3);
  size_t bn = id >> 2;
  int hw = (int)(bn % NN); int b = (int)(bn / NN);
  int y = hw >> 8, x = hw & 255;

  const float* ofs = offb + bn*96 + head*24;
  const float* awl = awb  + bn*48 + head*12;
  float wv[12]; float wmax=-1e30f;
  #pragma unroll
  for(int i=0;i<12;i++){ wv[i]=awl[i]; wmax=fmaxf(wmax,wv[i]); }
  float s=0.f;
  #pragma unroll
  for(int i=0;i<12;i++){ wv[i]=__expf(wv[i]-wmax); s+=wv[i]; }
  float inv = 1.f/s;

  float fbx = flow_b[(((size_t)b*2+0)*2+0)*NN+hw]*(1.f/WW);
  float fby = flow_b[(((size_t)b*2+0)*2+1)*NN+hw]*(1.f/HH);
  float ffx = flow_f[(((size_t)b*2+1)*2+0)*NN+hw]*(1.f/WW);
  float ffy = flow_f[(((size_t)b*2+1)*2+1)*NN+hw]*(1.f/HH);
  float refx = ((float)x+0.5f)*(1.f/WW), refy = ((float)y+0.5f)*(1.f/HH);

  float acc[16];
  #pragma unroll
  for(int d=0;d<16;d++) acc[d]=0.f;

  for(int l=0;l<NLEV;l++){
    float addx = (l==0)?fbx:((l==2)?ffx:0.f);
    float addy = (l==0)?fby:((l==2)?ffy:0.f);
    const bf16* vb = vsamp + ((((size_t)(b*TT+l))*NHEADS + head)*NN)*HEAD_DIM;
    for(int p=0;p<NPTS;p++){
      float lx = refx + ofs[(l*NPTS+p)*2+0]*(1.f/WW) + addx;
      float ly = refy + ofs[(l*NPTS+p)*2+1]*(1.f/HH) + addy;
      float gx = lx*WW - 0.5f, gy = ly*HH - 0.5f;
      float x0f=floorf(gx), y0f=floorf(gy);
      float wx=gx-x0f, wy=gy-y0f;
      int xi0=(int)x0f, yi0=(int)y0f;
      float ww = wv[l*NPTS+p]*inv;
      float cw[4] = { (1.f-wx)*(1.f-wy)*ww, wx*(1.f-wy)*ww, (1.f-wx)*wy*ww, wx*wy*ww };
      int cx[4]={xi0,xi0+1,xi0,xi0+1}, cy[4]={yi0,yi0,yi0+1,yi0+1};
      #pragma unroll
      for(int ci=0;ci<4;ci++){
        int xi=cx[ci], yi=cy[ci];
        if(xi>=0 && xi<WW && yi>=0 && yi<HH){
          const bf16* tp = vb + ((size_t)(yi*WW+xi))*HEAD_DIM;
          float w2 = cw[ci];
          #pragma unroll
          for(int d=0;d<16;d++) acc[d] += w2*(float)tp[d];
        }
      }
    }
  }
  bf16* orow = attn + bn*64 + head*16;
  #pragma unroll
  for(int d=0;d<16;d++) orow[d] = (bf16)acc[d];
}

// ---------------------------------------------------------------------------
extern "C" void kernel_launch(void* const* d_in, const int* in_sizes, int n_in,
                              void* d_out, int out_size, void* d_ws, size_t ws_size,
                              hipStream_t stream) {
  const float* frame_in  = (const float*)d_in[0];
  const float* srcframe  = (const float*)d_in[1];
  const float* flow_f    = (const float*)d_in[2];
  const float* flow_b    = (const float*)d_in[3];
  const float* prior     = (const float*)d_in[4];
  const float* map_w     = (const float*)d_in[5];
  const float* map_b     = (const float*)d_in[6];
  const float* emb_qk_w  = (const float*)d_in[7];
  const float* emb_qk_b  = (const float*)d_in[8];
  const float* emb_v_w   = (const float*)d_in[9];
  const float* emb_v_b   = (const float*)d_in[10];
  const float* off_w     = (const float*)d_in[11];
  const float* off_b     = (const float*)d_in[12];
  const float* aw_w      = (const float*)d_in[13];
  const float* aw_b      = (const float*)d_in[14];
  const float* vproj_w   = (const float*)d_in[15];
  const float* vproj_b   = (const float*)d_in[16];
  const float* outproj_w = (const float*)d_in[17];
  const float* outproj_b = (const float*)d_in[18];
  const float* ffn_w     = (const float*)d_in[19];
  const float* ffn_b     = (const float*)d_in[20];
  const float* fus_w     = (const float*)d_in[21];
  const float* fus_b     = (const float*)d_in[22];

  char* base = (char*)d_ws;
  size_t cur = 0;
  auto alignup = [](size_t v){ return (v + 255) & ~(size_t)255; };
  auto take = [&](size_t bytes)->char*{ char* p = base + cur; cur = alignup(cur + bytes); return p; };

  const size_t BN = (size_t)BB*NN;
  // Region X: {qkin(224ch)+fr192(192ch)} reused later by {off(f32,96)+aw(f32,48)+attn(bf16,64)}
  const size_t regX = BN*224*2 + BN*192*2;              // 109.05 MB (>= 92.3 MB alt use)
  char* pX = take(regX);
  bf16*  qkin  = (bf16*)pX;
  bf16*  fr192 = (bf16*)(pX + BN*224*2);
  float* offb  = (float*)pX;
  float* awb   = (float*)(pX + BN*96*4);
  bf16*  attn  = (bf16*)(pX + BN*96*4 + BN*48*4);
  // Region Y: valA reused later by ffnin + fusin
  char* pY = take((size_t)BB*TT*NN*64*2);               // 50.3 MB
  bf16*  valA  = (bf16*)pY;
  bf16*  ffnin = (bf16*)pY;
  bf16*  fusin = (bf16*)(pY + BN*64*2);
  bf16*  q     = (bf16*)take(BN*64*2);
  bf16*  vsamp = (bf16*)take((size_t)BB*TT*NN*64*2);
  bf16*  wqk   = (bf16*)take((size_t)64*9*224*2);
  bf16*  wv    = (bf16*)take((size_t)192*9*192*2);
  bf16*  wffn  = (bf16*)take((size_t)64*9*64*2);
  bf16*  wtv   = (bf16*)take((size_t)64*64*2);
  bf16*  wto   = (bf16*)take((size_t)96*64*2);
  bf16*  wta   = (bf16*)take((size_t)48*64*2);
  bf16*  wtf   = (bf16*)take((size_t)64*64*2);
  float* gate  = (float*)take((size_t)BB*CC*4);
  bf16*  wteff = (bf16*)take((size_t)BB*64*64*2);
  float* beff  = (float*)take((size_t)BB*64*4);
  char*  zpad  = take(4096);                            // zero page for halo lanes
  (void)ws_size; (void)n_in; (void)in_sizes; (void)out_size;

  (void)hipMemsetAsync(zpad, 0, 4096, stream);

  // 1) weight prepack + gating path
  k_prepack<<<2012, 256, 0, stream>>>(emb_qk_w, emb_v_w, ffn_w, vproj_w, off_w, aw_w, fus_w,
                                      wqk, wv, wffn, wtv, wto, wta, wtf);
  k_gate_mean<<<BB*CC, 256, 0, stream>>>(frame_in, prior, gate);
  k_select<<<1, 256, 0, stream>>>(gate, map_w, map_b, outproj_w, outproj_b, wteff, beff);

  // 2) warps + NHWC staging
  k_stage<<<(unsigned)(BN/256), 256, 0, stream>>>(frame_in, srcframe, flow_f, flow_b, qkin, fr192);

  // 3) WMMA convs (TDM + async LDS double-buffered A tiles)
  k_conv3x3<224,0><<<dim3(1024,1), 256, 0, stream>>>(qkin,  wqk,  emb_qk_b, q,     nullptr, zpad);
  k_conv3x3<192,1><<<dim3(1024,3), 256, 0, stream>>>(fr192, wv,   emb_v_b,  valA,  nullptr, zpad);

  // 4) WMMA GEMMs: vproj, offsets, attention-weight logits
  k_gemm64<4,10><<<3072, 256, 0, stream>>>(valA, wtv, vproj_b, vsamp);
  k_gemm64<6,11><<<1024, 256, 0, stream>>>(q,    wto, off_b,   offb);
  k_gemm64<3,12><<<1024, 256, 0, stream>>>(q,    wta, aw_b,    awb);

  // 5) deformable sampling
  k_sample<<<(unsigned)(BB*NN*NHEADS/256), 256, 0, stream>>>(offb, awb, vsamp, flow_f, flow_b, attn);

  // 6) outproj (folded select), ffn conv, fus 1x1 + lrelu -> output
  k_gemm64<4,13><<<1024, 256, 0, stream>>>(attn, wteff, beff, ffnin);
  k_conv3x3<64,2><<<dim3(1024,1), 256, 0, stream>>>(ffnin, wffn, ffn_b, fusin, frame_in, zpad);
  k_gemm64<4,14><<<1024, 256, 0, stream>>>(fusin, wtf, fus_b, d_out);
}
